// Attention_36472862277692
// MI455X (gfx1250) — matrix operand order repaired and checked by reading, not compiled, not run on hardware
//
#include <hip/hip_runtime.h>
#include <stdint.h>

#define HH 8
#define BB 4
#define LL 2048
#define SS 2048
#define DD 1024
#define PP 1024
#define EE 128   // head dim = P/H

typedef __attribute__((ext_vector_type(16))) __bf16 v16bf;
typedef __attribute__((ext_vector_type(8)))  float  v8f;
typedef unsigned int v4u __attribute__((ext_vector_type(4)));
typedef int          v8i __attribute__((ext_vector_type(8)));
typedef int          v4i __attribute__((ext_vector_type(4)));

union FragBF { unsigned short u[16]; v16bf v; };

__device__ __forceinline__ unsigned short f2bf(float f) {
  __bf16 h = (__bf16)f;                         // native v_cvt (RNE)
  return __builtin_bit_cast(unsigned short, h);
}

// DPP butterfly step within each 16-lane half-row (no LDS, no waits).
// ctrl: 0xB1 = quad_perm(1,0,3,2) -> xor 1 ; 0x4E = quad_perm(2,3,0,1) -> xor 2
//       0x164 = row_xmask:4 -> xor 4       ; 0x168 = row_xmask:8 -> xor 8
template<int CTRL>
__device__ __forceinline__ float dpp_xor(float x) {
  int i = __builtin_bit_cast(int, x);
  int r = __builtin_amdgcn_update_dpp(i, i, CTRL, 0xF, 0xF, true);
  return __builtin_bit_cast(float, r);
}
__device__ __forceinline__ float row16_max(float x) {
  x = fmaxf(x, dpp_xor<0xB1>(x));
  x = fmaxf(x, dpp_xor<0x4E>(x));
  x = fmaxf(x, dpp_xor<0x164>(x));
  x = fmaxf(x, dpp_xor<0x168>(x));
  return x;
}
__device__ __forceinline__ float row16_sum(float x) {
  x += dpp_xor<0xB1>(x);
  x += dpp_xor<0x4E>(x);
  x += dpp_xor<0x164>(x);
  x += dpp_xor<0x168>(x);
  return x;
}

// ---------------------------------------------------------------------------
// TDM: DMA a (rows x 128) bf16 tile (row stride 1024 elements in memory) into
// LDS, padding each 256B row with 8B so the LDS row stride is 132 u16.
// D# layout per CDNA5 ISA. 6-arg clang builtin form.
// Issue from ONE wave only (EXEC ignored; avoids duplicate DMAs).
// ---------------------------------------------------------------------------
__device__ __forceinline__ void tdm_load_tile128(const unsigned short* gptr,
                                                 const unsigned short* ldsptr,
                                                 unsigned int rows)
{
  uint64_t ga = (uint64_t)(uintptr_t)gptr;
  uint32_t la = (uint32_t)(uintptr_t)ldsptr;    // low 32 bits = LDS offset

  v4u g0;
  g0[0] = 1u;                                    // count=1, user descriptor
  g0[1] = la;                                    // lds_addr
  g0[2] = (uint32_t)ga;                          // global_addr[31:0]
  g0[3] = (uint32_t)((ga >> 32) & 0x01FFFFFFu)   // global_addr[56:32]
        | (2u << 30);                            // type = 2 ("image")

  v8i g1;
  g1[0] = (int)((1u << 16)      // data_size = 1 -> 2 bytes
              | (1u << 20)      // pad_enable
              | (5u << 22)      // pad_interval = 5 -> every 64 DWORDs (256B row)
              | (1u << 25));    // pad_amount = 1 -> 2 DWORDs (8B) pad
  g1[1] = (int)(128u << 16);    // tensor_dim0 = 128 (low16 at [31:16])
  g1[2] = (int)(rows << 16);    // tensor_dim0 hi = 0 ; tensor_dim1 low16
  g1[3] = (int)(128u << 16);    // tensor_dim1 hi = 0 ; tile_dim0 = 128
  g1[4] = (int)rows;            // tile_dim1 = rows ; tile_dim2 = 0
  g1[5] = (int)1024u;           // tensor_dim0_stride = 1024 elements
  g1[6] = 0;
  g1[7] = 0;

  v4i z4 = {0, 0, 0, 0};        // <=2D tensor: groups 2/3 unused
  v8i z8 = {0, 0, 0, 0, 0, 0, 0, 0};
  __builtin_amdgcn_tensor_load_to_lds(g0, g1, z4, z4, z8, 0);
}

// ---------------------------------------------------------------------------
// 128x128 tile GEMM: C[M,N] = A[M,K] @ W[K,N] + bias, bf16 WMMA, f32 accum.
// 256 threads = 8 waves in a 2x4 grid; each wave owns 64x32 (4x2 fragments):
// 8 WMMAs per 6 fragment loads per K=32 step.
// A_F32: A is fp32 (converted while staging) else bf16. C_F32: fp32 vs bf16 out.
// ---------------------------------------------------------------------------
template<bool A_F32, bool C_F32>
__global__ __launch_bounds__(256)
void gemm128x128(const void* __restrict__ Ap, const float* __restrict__ Wp,
                 const float* __restrict__ bias, void* __restrict__ Cp,
                 int M, int N, int K)
{
  constexpr int AST = 36;                        // padded A-tile row stride
  __shared__ unsigned short Alds[128 * AST];
  __shared__ unsigned short Blds[32 * 128];

  const int tid  = threadIdx.x;
  const int lane = tid & 31;
  const int wave = tid >> 5;
  const int half = lane >> 4;
  const int l15  = lane & 15;

  const int m0 = blockIdx.y * 128;
  const int n0 = blockIdx.x * 128;
  const int wm = (wave >> 2) * 64;               // wave row: 0 or 64
  const int wn = (wave & 3) * 32;                // wave col: 0,32,64,96

  v8f acc[4][2] = {};

  for (int k0 = 0; k0 < K; k0 += 32) {
    // stage A tile 128x32 (32 lanes cover one 32-wide row)
    #pragma unroll
    for (int e = 0; e < 16; ++e) {
      int idx = e * 256 + tid;
      int r = idx >> 5, c = idx & 31;
      unsigned short v;
      if (A_F32) v = f2bf(((const float*)Ap)[(size_t)(m0 + r) * K + (k0 + c)]);
      else       v = ((const unsigned short*)Ap)[(size_t)(m0 + r) * K + (k0 + c)];
      Alds[r * AST + c] = v;
    }
    // stage W tile 32x128 (fp32 weights -> bf16)
    #pragma unroll
    for (int e = 0; e < 16; ++e) {
      int idx = e * 256 + tid;
      int r = idx >> 7, c = idx & 127;
      Blds[r * 128 + c] = f2bf(Wp[(size_t)(k0 + r) * N + (n0 + c)]);
    }
    __syncthreads();

    FragBF a[4], b[2];
    #pragma unroll
    for (int i = 0; i < 4; ++i) {
      int am = wm + i * 16 + l15;
      #pragma unroll
      for (int t = 0; t < 16; ++t) {
        int kk = t + 16 * half;                            // contiguous k order (matches B)
        a[i].u[t] = Alds[am * AST + kk];
      }
    }
    #pragma unroll
    for (int j = 0; j < 2; ++j) {
      int bn = wn + j * 16 + l15;
      #pragma unroll
      for (int t = 0; t < 16; ++t) {
        int kk = t + 16 * half;                            // contiguous k order (matches A)
        b[j].u[t] = Blds[kk * 128 + bn];
      }
    }
    #pragma unroll
    for (int i = 0; i < 4; ++i)
      #pragma unroll
      for (int j = 0; j < 2; ++j)
        acc[i][j] = __builtin_amdgcn_wmma_f32_16x16x32_bf16(
            false, a[i].v, false, b[j].v, (short)0, acc[i][j], false, false);

    __syncthreads();
  }

  // epilogue: C layout row = r + 8*half, col = lane&15
  #pragma unroll
  for (int i = 0; i < 4; ++i) {
    #pragma unroll
    for (int j = 0; j < 2; ++j) {
      #pragma unroll
      for (int r = 0; r < 8; ++r) {
        int row = m0 + wm + i * 16 + r + 8 * half;
        int col = n0 + wn + j * 16 + l15;
        float v = acc[i][j][r] + bias[col];
        if (C_F32) ((float*)Cp)[(size_t)row * N + col] = v;
        else ((unsigned short*)Cp)[(size_t)row * N + col] = f2bf(v);
      }
    }
  }
}

// ---------------------------------------------------------------------------
// Flash attention: block = (64 L-rows) x one (b,h); 4 waves, wave owns 16 rows.
// Q/K/V tiles DMAed into LDS by the Tensor Data Mover (wave 0 issues),
// K/V double-buffered: chunk i+1 streams in while chunk i is computed.
// Online softmax (DPP row reductions), O accumulated fp32 in WMMA C fragments.
// ---------------------------------------------------------------------------
__global__ __launch_bounds__(128)
void flash_attn(const unsigned short* __restrict__ Qp,
                const unsigned short* __restrict__ Kp,
                const unsigned short* __restrict__ Vp,
                unsigned short* __restrict__ ctx)
{
  constexpr int KST = 132;                       // 128 data + 4 pad u16 (TDM pad)
  __shared__ unsigned short Qlds[64 * KST];
  __shared__ unsigned short Klds[2][32 * KST];
  __shared__ unsigned short Vlds[2][32 * KST];
  __shared__ unsigned short Plds[4][16 * 36];    // per-wave P transpose buffer

  const int tid  = threadIdx.x;
  const int lane = tid & 31;
  const int wave = tid >> 5;
  const int half = lane >> 4;
  const int l15  = lane & 15;

  const int bh = blockIdx.y;
  const int b  = bh / HH;
  const int h  = bh % HH;
  const int m0 = blockIdx.x * 64;

  const float scale = 0.08838834764831845f;      // 1/sqrt(128)
  const int   NIT   = SS / 32;

  // ---- prologue: TDM Q (64x128) + first K/V chunk (32x128 each) ----
  if (wave == 0) {
    tdm_load_tile128(Qp + (size_t)(b * LL + m0) * PP + h * EE, Qlds, 64);
    size_t base0 = (size_t)(b * SS) * PP + h * EE;
    tdm_load_tile128(Kp + base0, Klds[0], 32);
    tdm_load_tile128(Vp + base0, Vlds[0], 32);
    __builtin_amdgcn_s_wait_tensorcnt(2);        // Q complete (in-order)
  }
  __syncthreads();

  // ---- Q fragments for the 4 K=32 chunks of E=128, kept in registers ----
  FragBF qf[4];
  {
    int qm = wave * 16 + l15;
    #pragma unroll
    for (int kc = 0; kc < 4; ++kc)
      #pragma unroll
      for (int t = 0; t < 16; ++t) {
        int kk = kc * 32 + t + 16 * half;
        qf[kc].u[t] = Qlds[qm * KST + kk];
      }
  }

  v8f o[8] = {};
  float mrun[8], lrun[8];
  #pragma unroll
  for (int r = 0; r < 8; ++r) { mrun[r] = -1e30f; lrun[r] = 0.f; }

  for (int it = 0; it < NIT; ++it) {
    const int p = it & 1;
    __syncthreads();                             // buf p^1 free for overwrite
    if (wave == 0) {
      if (it + 1 < NIT) {                        // prefetch chunk it+1
        size_t nb = (size_t)(b * SS + (it + 1) * 32) * PP + h * EE;
        tdm_load_tile128(Kp + nb, Klds[p ^ 1], 32);
        tdm_load_tile128(Vp + nb, Vlds[p ^ 1], 32);
        __builtin_amdgcn_s_wait_tensorcnt(2);    // chunk it complete
      } else {
        __builtin_amdgcn_s_wait_tensorcnt(0);
      }
    }
    __syncthreads();                             // publish chunk it to all waves

    // ---- scores: S[l][s] = sum_e Q[l][e] * K[s][e] ----
    v8f sc[2] = {};
    #pragma unroll
    for (int nt = 0; nt < 2; ++nt) {
      int srow = nt * 16 + l15;                  // B-frag column n = local s
      #pragma unroll
      for (int kc = 0; kc < 4; ++kc) {
        FragBF kb;
        #pragma unroll
        for (int t = 0; t < 16; ++t)
          kb.u[t] = Klds[p][srow * KST + kc * 32 + t + 16 * half];
        sc[nt] = __builtin_amdgcn_wmma_f32_16x16x32_bf16(
            false, qf[kc].v, false, kb.v, (short)0, sc[nt], false, false);
      }
    }

    // ---- online softmax over the 32-wide chunk (DPP reductions) ----
    #pragma unroll
    for (int r = 0; r < 8; ++r) {
      float a  = sc[0][r] * scale;
      float b2 = sc[1][r] * scale;
      float mx = row16_max(fmaxf(a, b2));        // uniform within 16-lane group
      float mnew = fmaxf(mrun[r], mx);
      float corr = __expf(mrun[r] - mnew);
      mrun[r] = mnew;
      float p0 = __expf(a - mnew);
      float p1 = __expf(b2 - mnew);
      float sum = row16_sum(p0 + p1);
      lrun[r] = lrun[r] * corr + sum;
      sc[0][r] = p0; sc[1][r] = p1;
      #pragma unroll
      for (int f = 0; f < 8; ++f) o[f][r] *= corr;
    }

    // ---- transpose P (C-layout -> A-layout) through per-wave LDS ----
    #pragma unroll
    for (int nt = 0; nt < 2; ++nt)
      #pragma unroll
      for (int r = 0; r < 8; ++r)
        Plds[wave][(r + 8 * half) * 36 + nt * 16 + l15] = f2bf(sc[nt][r]);

    __syncthreads();

    // ---- O += P @ V ----
    FragBF pf;
    #pragma unroll
    for (int t = 0; t < 16; ++t) {
      int kk = t + 16 * half;
      pf.u[t] = Plds[wave][l15 * 36 + kk];
    }
    #pragma unroll
    for (int f = 0; f < 8; ++f) {
      FragBF vb;
      #pragma unroll
      for (int t = 0; t < 16; ++t)
        vb.u[t] = Vlds[p][(t + 16 * half) * KST + f * 16 + l15];
      o[f] = __builtin_amdgcn_wmma_f32_16x16x32_bf16(
          false, pf.v, false, vb.v, (short)0, o[f], false, false);
    }
  }

  // ---- normalize and store ctx (bf16) ----
  #pragma unroll
  for (int r = 0; r < 8; ++r) {
    float inv = 1.0f / lrun[r];
    int row = m0 + wave * 16 + r + 8 * half;
    #pragma unroll
    for (int f = 0; f < 8; ++f) {
      int col = h * EE + f * 16 + l15;
      ctx[(size_t)(b * LL + row) * PP + col] = f2bf(o[f][r] * inv);
    }
  }
}

// ---------------------------------------------------------------------------
extern "C" void kernel_launch(void* const* d_in, const int* in_sizes, int n_in,
                              void* d_out, int out_size, void* d_ws, size_t ws_size,
                              hipStream_t stream) {
  (void)in_sizes; (void)n_in; (void)out_size; (void)ws_size;
  const float* queries = (const float*)d_in[0];
  const float* keys    = (const float*)d_in[1];
  const float* values  = (const float*)d_in[2];
  const float* Wq      = (const float*)d_in[3];
  const float* bq      = (const float*)d_in[4];
  const float* Wk      = (const float*)d_in[5];
  const float* bk      = (const float*)d_in[6];
  const float* Wv      = (const float*)d_in[7];
  const float* bv      = (const float*)d_in[8];
  const float* Wo      = (const float*)d_in[9];
  const float* bo      = (const float*)d_in[10];
  float* out = (float*)d_out;

  const int M = BB * LL;                        // 8192 rows
  size_t qkv = (size_t)M * PP;                  // 8M bf16 elems each
  unsigned short* Qp  = (unsigned short*)d_ws;
  unsigned short* Kp  = Qp + qkv;
  unsigned short* Vp  = Kp + qkv;
  unsigned short* ctx = Vp + qkv;               // 64 MB total in d_ws

  dim3 blk(256);
  dim3 gproj(PP / 128, M / 128);
  gemm128x128<true, false><<<gproj, blk, 0, stream>>>(queries, Wq, bq, Qp, M, PP, DD);
  gemm128x128<true, false><<<gproj, blk, 0, stream>>>(keys,    Wk, bk, Kp, M, PP, DD);
  gemm128x128<true, false><<<gproj, blk, 0, stream>>>(values,  Wv, bv, Vp, M, PP, DD);

  dim3 ablk(128);
  dim3 gattn(LL / 64, BB * HH);
  flash_attn<<<gattn, ablk, 0, stream>>>(Qp, Kp, Vp, ctx);

  dim3 gout(DD / 128, M / 128);
  gemm128x128<false, true><<<gout, blk, 0, stream>>>(ctx, Wo, bo, out, M, DD, PP);
}
